// Attention_18777597018238
// MI455X (gfx1250) — matrix operand order repaired and checked by reading, not compiled, not run on hardware
//
#include <hip/hip_runtime.h>
#include <hip/hip_bf16.h>

typedef __attribute__((ext_vector_type(16))) __bf16 v16bf;
typedef __attribute__((ext_vector_type(8)))  __bf16 v8bf;
typedef __attribute__((ext_vector_type(8)))  float  v8f;
typedef __attribute__((ext_vector_type(4)))  unsigned int u32x4;
typedef __attribute__((ext_vector_type(8)))  int i32x8;
typedef __attribute__((ext_vector_type(4)))  int i32x4;

union Frag16 { v16bf v; v8bf h[2]; };

constexpr int kB   = 8;
constexpr int kC   = 192;
constexpr int kC3  = 576;
constexpr int kHC  = 48;
constexpr int kN   = 16384;   // 128*128
constexpr float kEPS = 1e-12f;

// ---------------------------------------------------------------------------
// TDM helper: issue a 2-D tensor_load_to_lds of a [rows x 32] bf16 tile whose
// row stride in the source tensor is `strideElems` (bf16 elements).
// D# layout per CDNA5 ISA ch.8 (group0: count/lds/global/type, group1: dims).
// Caller must be a single wave; waits TENSORcnt==0 before returning.
// ---------------------------------------------------------------------------
__device__ __forceinline__ void tdm_load_tile_bf16(
    const __bf16* gsrc, unsigned ldsOff, int rows, int strideElems) {
  unsigned long long ga = (unsigned long long)(size_t)gsrc;
  u32x4 g0 = { 1u,                                   // count=1, user mode
               ldsOff,                               // lds_addr (bytes)
               (unsigned)ga,                         // global_addr[31:0]
               ((unsigned)(ga >> 32) & 0x01FFFFFFu)  // global_addr[56:32]
                   | (2u << 30) };                   // type=2 ("image")
  i32x8 g1 = { 0x00010000,            // wg_mask=0, data_size=1 (2 bytes)
               (32 << 16),            // tensor_dim0 = 32 (lo16)
               (rows << 16),          // tensor_dim0 hi=0 | tensor_dim1 lo16
               (32 << 16),            // tensor_dim1 hi=0 | tile_dim0 = 32
               rows,                  // tile_dim1 | tile_dim2=0
               strideElems,           // tensor_dim0_stride[31:0]
               0, 0 };
  i32x4 g2 = { 0, 0, 0, 0 };
  i32x4 g3 = { 0, 0, 0, 0 };
#if __clang_major__ >= 23
  i32x8 g4 = { 0, 0, 0, 0, 0, 0, 0, 0 };
  __builtin_amdgcn_tensor_load_to_lds(g0, g1, g2, g3, g4, 0);
#else
  __builtin_amdgcn_tensor_load_to_lds(g0, g1, g2, g3, 0);
#endif
  __builtin_amdgcn_s_wait_tensorcnt(0);
}

// ---------------------------------------------------------------------------
// K0: convert weights f32 -> bf16
// ---------------------------------------------------------------------------
__global__ __launch_bounds__(256) void cvt_weights_kernel(
    const float* __restrict__ wq, const float* __restrict__ wp,
    __bf16* __restrict__ wqb, __bf16* __restrict__ wpb) {
  int i = blockIdx.x * 256 + threadIdx.x;
  if (i < kC3 * kC) wqb[i] = (__bf16)wq[i];
  if (i < kC * kC)  wpb[i] = (__bf16)wp[i];
}

// ---------------------------------------------------------------------------
// Tiled WMMA GEMM: C[b][M][N] = A[M][K](bf16) * B[b][K][N](SRC)
// Block tile: 64(M) x 128(N), 8 waves, each wave one 64x16 column strip.
// A tile (row-major copy) staged by the Tensor Data Mover from wave 0;
// B tile staged by all threads (f32->bf16 convert + transpose into
// lsB[col*32+k] so a B fragment is one contiguous 32-byte LDS read).
// ---------------------------------------------------------------------------
template <typename SRC>
__global__ __launch_bounds__(256) void gemm_wmma_kernel(
    const __bf16* __restrict__ A, const SRC* __restrict__ Bm,
    float* __restrict__ Cm, int M, int K, int N,
    long strideB, long strideC) {
  __shared__ __bf16 lsA[64 * 32];   // 4 KB
  __shared__ __bf16 lsB[128 * 32];  // 8 KB

  const int tid = threadIdx.x;
  const int p0 = blockIdx.x * 128;
  const int m0 = blockIdx.y * 64;
  const long bb = blockIdx.z;
  const SRC* Bp = Bm + bb * strideB;
  float* Cp = Cm + bb * strideC;

  const int w = tid >> 5, l = tid & 31;
  const int n = l & 15;
  const int kbA = (l < 16) ? 0 : 16;   // A half-lane K offset
  const int kbB = (l < 16) ? 0 : 16;   // B half-lane K offset

  const unsigned lsAoff = (unsigned)(size_t)&lsA[0];

  v8f acc[4] = {};

  for (int k0 = 0; k0 < K; k0 += 32) {
    // stage A via TDM: 64 rows x 32 K, row stride K, issued by wave 0 only
    if (tid < 32) {
      tdm_load_tile_bf16(A + (long)m0 * K + k0, lsAoff, 64, K);
    }
    // stage B: 32 K-rows x 128 cols, transposed into lsB[col][k]
    {
      int r = tid >> 3, seg = (tid & 7) * 16;
      const SRC* src = Bp + (long)(k0 + r) * N + p0 + seg;
      __builtin_prefetch(Bp + (long)(k0 + 32 + r) * N + p0 + seg, 0, 1);
#pragma unroll
      for (int i = 0; i < 16; ++i)
        lsB[(seg + i) * 32 + r] = (__bf16)(float)src[i];
    }
    __syncthreads();

    Frag16 bf;
    bf.v = *(const v16bf*)&lsB[(w * 16 + n) * 32 + kbB];
#pragma unroll
    for (int a = 0; a < 4; ++a) {
      Frag16 af;
      af.h[0] = *(const v8bf*)&lsA[(a * 16 + n) * 32 + kbA];
      af.h[1] = *(const v8bf*)&lsA[(a * 16 + n) * 32 + kbA + 8];
      acc[a] = __builtin_amdgcn_wmma_f32_16x16x32_bf16(
          false, af.v, false, bf.v, (short)0, acc[a], false, false);
    }
    __syncthreads();
  }

  const int mhalf = (l < 16) ? 0 : 8;
#pragma unroll
  for (int a = 0; a < 4; ++a) {
#pragma unroll
    for (int r = 0; r < 8; ++r) {
      int Mg = m0 + a * 16 + mhalf + r;
      int Ng = p0 + w * 16 + n;
      Cp[(long)Mg * N + Ng] = acc[a][r];
    }
  }
}

// ---------------------------------------------------------------------------
// K2: depthwise 3x3 conv, SAME padding; writes bf16 for later matrix stages
// ---------------------------------------------------------------------------
__global__ __launch_bounds__(256) void dwconv_kernel(
    const float* __restrict__ in, const float* __restrict__ wdw,
    __bf16* __restrict__ out) {
  long idx = (long)blockIdx.x * 256 + threadIdx.x;
  const long total = (long)kB * kC3 * kN;
  if (idx >= total) return;
  int p = (int)(idx & (kN - 1));
  long bc = idx >> 14;              // b*576 + o
  int o = (int)(bc % kC3);
  int y = p >> 7, x = p & 127;
  const float* ip = in + (bc << 14);
  const float* wt = wdw + o * 9;
  float s = 0.f;
#pragma unroll
  for (int ky = 0; ky < 3; ++ky) {
    int yy = y + ky - 1;
    if (yy < 0 || yy >= 128) continue;
#pragma unroll
    for (int kx = 0; kx < 3; ++kx) {
      int xx = x + kx - 1;
      if (xx < 0 || xx >= 128) continue;
      s += ip[yy * 128 + xx] * wt[ky * 3 + kx];
    }
  }
  out[idx] = (__bf16)s;
}

// ---------------------------------------------------------------------------
// K3: L2 norms of q and k rows (length 16384), one block per row
// ---------------------------------------------------------------------------
__global__ __launch_bounds__(256) void norm_kernel(
    const __bf16* __restrict__ qkv, float* __restrict__ nq,
    float* __restrict__ nk) {
  __shared__ float red[256];
  int rid = blockIdx.x;             // 0..3071
  int isK = rid >= 1536;
  int r = isK ? rid - 1536 : rid;   // b*192 + hd*48 + ch
  int b = r / kC, ch = r % kC;
  const __bf16* row = qkv + ((long)b * kC3 + (isK ? kC : 0) + ch) * (long)kN;
  float s = 0.f;
  for (int i = threadIdx.x; i < kN; i += 256) {
    float v = (float)row[i];
    s += v * v;
  }
  red[threadIdx.x] = s;
  __syncthreads();
  for (int st = 128; st > 0; st >>= 1) {
    if (threadIdx.x < st) red[threadIdx.x] += red[threadIdx.x + st];
    __syncthreads();
  }
  if (threadIdx.x == 0) (isK ? nk : nq)[r] = sqrtf(red[0]);
}

// ---------------------------------------------------------------------------
// K4: attn = softmax( (q.k^T) * tau / (|q||k|) ), one block per (b,head),
// 9 waves each computing a 16x16 tile via 512 k32 WMMAs (operands straight
// from global: rows of q/k are contiguous 32 KB).
// ---------------------------------------------------------------------------
__global__ __launch_bounds__(288) void attn_kernel(
    const __bf16* __restrict__ qkv, const float* __restrict__ nq,
    const float* __restrict__ nk, const float* __restrict__ temp,
    float* __restrict__ attn) {
  __shared__ float sA[48 * 48];
  const int bh = blockIdx.x;        // b*4 + hd
  const int b = bh >> 2, hd = bh & 3;
  const int tid = threadIdx.x;
  const int w = tid >> 5, l = tid & 31;
  const int mi = w / 3, ni = w % 3;
  const int m = l & 15;
  const int kbA = (l < 16) ? 0 : 16;
  const int kbB = (l < 16) ? 0 : 16;

  const __bf16* qrow = qkv + ((long)b * kC3 + hd * kHC + mi * 16 + m) * (long)kN;
  const __bf16* krow = qkv + ((long)b * kC3 + kC + hd * kHC + ni * 16 + m) * (long)kN;

  v8f acc = {};
  for (int k0 = 0; k0 < kN; k0 += 32) {
    __builtin_prefetch(qrow + k0 + 512, 0, 1);
    __builtin_prefetch(krow + k0 + 512, 0, 1);
    Frag16 af, bf;
    af.h[0] = *(const v8bf*)(qrow + k0 + kbA);
    af.h[1] = *(const v8bf*)(qrow + k0 + kbA + 8);
    bf.v    = *(const v16bf*)(krow + k0 + kbB);
    acc = __builtin_amdgcn_wmma_f32_16x16x32_bf16(
        false, af.v, false, bf.v, (short)0, acc, false, false);
  }
  const int mhalf = (l < 16) ? 0 : 8;
#pragma unroll
  for (int r = 0; r < 8; ++r)
    sA[(mi * 16 + mhalf + r) * 48 + ni * 16 + m] = acc[r];
  __syncthreads();

  if (tid < 48) {  // fused normalize-scale + softmax, one thread per row
    const int c = tid;
    const float tval = temp[hd];
    const float qn = fmaxf(nq[bh * 48 + c], kEPS);
    float row[48];
    float mx = -1e30f;
#pragma unroll 1
    for (int d = 0; d < 48; ++d) {
      float kn = fmaxf(nk[bh * 48 + d], kEPS);
      float v = sA[c * 48 + d] * tval / (qn * kn);
      row[d] = v;
      mx = fmaxf(mx, v);
    }
    float sum = 0.f;
#pragma unroll 1
    for (int d = 0; d < 48; ++d) { row[d] = __expf(row[d] - mx); sum += row[d]; }
    float inv = 1.f / sum;
#pragma unroll 1
    for (int d = 0; d < 48; ++d)
      attn[((long)bh * 48 + c) * 48 + d] = row[d] * inv;
  }
}

// ---------------------------------------------------------------------------
// K5: out = attn(48x48) @ v(48x16384) per (b,head). K=48 is tiny and this
// stage is memory-bound, so VALU with attn in LDS; coalesced v reads.
// attn -> LDS staging uses async global->LDS copies (ASYNCcnt path).
// ---------------------------------------------------------------------------
__global__ __launch_bounds__(256) void av_kernel(
    const __bf16* __restrict__ qkv, const float* __restrict__ attn,
    __bf16* __restrict__ outa) {
  __shared__ float sA[48 * 48];     // 9216 B = 576 x 16 B
  const int bh = blockIdx.y, b = bh >> 2, hd = bh & 3;
  const int tid = threadIdx.x;

  {
    const unsigned ldsBase = (unsigned)(size_t)&sA[0];
    unsigned long long gBase =
        (unsigned long long)(size_t)(attn + (long)bh * 2304);
    for (int i = tid; i < 576; i += 256) {
      unsigned ldsOff = ldsBase + i * 16u;
      unsigned long long ga = gBase + (unsigned long long)i * 16u;
      asm volatile("global_load_async_to_lds_b128 %0, %1, off"
                   :: "v"(ldsOff), "v"(ga) : "memory");
    }
    asm volatile("s_wait_asynccnt 0x0" ::: "memory");
  }
  __syncthreads();

  const __bf16* vbase = qkv + ((long)b * kC3 + 2 * kC + hd * kHC) * (long)kN;
  __bf16* obase = outa + ((long)b * kC + hd * kHC) * (long)kN;
  const int p0 = blockIdx.x * 2048;
  for (int rep = 0; rep < 8; ++rep) {
    int p = p0 + rep * 256 + tid;
    float vv[48];
#pragma unroll
    for (int d = 0; d < 48; ++d) vv[d] = (float)vbase[(long)d * kN + p];
#pragma unroll 1
    for (int c = 0; c < 48; ++c) {
      float s = 0.f;
#pragma unroll
      for (int d = 0; d < 48; ++d) s += sA[c * 48 + d] * vv[d];
      obase[(long)c * kN + p] = (__bf16)s;
    }
  }
}

// ---------------------------------------------------------------------------
extern "C" void kernel_launch(void* const* d_in, const int* in_sizes, int n_in,
                              void* d_out, int out_size, void* d_ws,
                              size_t ws_size, hipStream_t stream) {
  const float* x     = (const float*)d_in[0];
  const float* w_qkv = (const float*)d_in[1];
  const float* w_dw  = (const float*)d_in[2];
  const float* temp  = (const float*)d_in[3];
  const float* w_prj = (const float*)d_in[4];
  float* out = (float*)d_out;

  char* ws = (char*)d_ws;
  size_t off = 0;
  auto take = [&](size_t bytes) -> void* {
    void* p = ws + off;
    off = (off + bytes + 255) & ~(size_t)255;
    return p;
  };
  float*  qkv_f  = (float*) take((size_t)kB * kC3 * kN * 4);   // 302 MB
  __bf16* qkv_dw = (__bf16*)take((size_t)kB * kC3 * kN * 2);   // 151 MB
  float*  nq     = (float*) take(1536 * 4);
  float*  nk     = (float*) take(1536 * 4);
  float*  attn   = (float*) take((size_t)32 * 48 * 48 * 4);
  __bf16* wq_b   = (__bf16*)take((size_t)kC3 * kC * 2);
  __bf16* wp_b   = (__bf16*)take((size_t)kC * kC * 2);
  __bf16* outa   = (__bf16*)take((size_t)kB * kC * kN * 2);    // 50 MB
  (void)ws_size; (void)in_sizes; (void)n_in; (void)out_size;

  // K0: weights -> bf16
  cvt_weights_kernel<<<(kC3 * kC + 255) / 256, 256, 0, stream>>>(
      w_qkv, w_prj, wq_b, wp_b);

  // K1: qkv = W_qkv(576x192) @ x(192x16384) per batch  (WMMA bf16 + TDM)
  gemm_wmma_kernel<float><<<dim3(kN / 128, kC3 / 64, kB), 256, 0, stream>>>(
      wq_b, x, qkv_f, kC3, kC, kN, (long)kC * kN, (long)kC3 * kN);

  // K2: depthwise 3x3
  long dwTotal = (long)kB * kC3 * kN;
  dwconv_kernel<<<(int)((dwTotal + 255) / 256), 256, 0, stream>>>(
      qkv_f, w_dw, qkv_dw);

  // K3: row norms for q and k
  norm_kernel<<<3072, 256, 0, stream>>>(qkv_dw, nq, nk);

  // K4: channel attention matrix + softmax  (WMMA bf16)
  attn_kernel<<<32, 288, 0, stream>>>(qkv_dw, nq, nk, temp, attn);

  // K5: attn @ v  (async global->LDS staging)
  av_kernel<<<dim3(8, 32), 256, 0, stream>>>(qkv_dw, attn, outa);

  // K6: projection GEMM  (WMMA bf16 + TDM)
  gemm_wmma_kernel<__bf16><<<dim3(kN / 128, kC / 64, kB), 256, 0, stream>>>(
      wp_b, outa, out, kC, kC, kN, (long)kC * kN, (long)kC * kN);
}
